// SelfExpandingGate_57329223467223
// MI455X (gfx1250) — matrix operand order repaired and checked by reading, not compiled, not run on hardware
//
#include <hip/hip_runtime.h>
#include <hip/hip_bf16.h>
#include <math.h>

typedef __bf16 bf16;
typedef bf16  bf16x16 __attribute__((ext_vector_type(16)));
typedef float f32x8   __attribute__((ext_vector_type(8)));

#define XW 320            // row width of x  (128 s + 64*3 v)
#define OW 320            // row width of out (128 scal + 192 gated)
#define KEPS 1e-9f

// bf16 weight workspace layout (element counts)
constexpr size_t N000 = (size_t)16384 * 192;   // (s⊗s)  K=16384, N=192
constexpr size_t N110 = (size_t)4096  * 192;   // (v·vT) K=4096,  N=192
constexpr size_t N011 = (size_t)8192  * 64;    // (s⊗v)  K=8192,  N=64
constexpr size_t N101 = (size_t)8192  * 64;    // (v⊗s)  K=8192,  N=64
constexpr size_t OFF000 = 0;
constexpr size_t OFF110 = OFF000 + N000;
constexpr size_t OFF011 = OFF110 + N110;
constexpr size_t OFF101 = OFF011 + N011;
constexpr size_t WTOT   = OFF101 + N101;       // 4,980,736 bf16 = ~9.5 MB

__device__ __forceinline__ f32x8 wmma_bf16(bf16x16 a, bf16x16 b, f32x8 c) {
    return __builtin_amdgcn_wmma_f32_16x16x32_bf16(false, a, false, b, (short)0, c, false, false);
}

// ---------------------------------------------------------------------------
// Prep: fp32 weights -> bf16 in per-lane WMMA B-fragment order.
// Element o within a region:  o = ((kb*NT + nt)*32 + lane)*16 + e
//   lanes 0-15 : K = kb*32 + (e&7) + 16*(e>>3)      (N = nt*16 + lane)
//   lanes 16-31: K = kb*32 + 8 + (e&7) + 16*(e>>3) (N = nt*16 + lane-16)
// ---------------------------------------------------------------------------
__global__ __launch_bounds__(256) void prep_weights(
    const float* __restrict__ W000a, const float* __restrict__ W000b,
    const float* __restrict__ W110a, const float* __restrict__ W110b,
    const float* __restrict__ W011,  const float* __restrict__ W101,
    bf16* __restrict__ ws)
{
    size_t i = (size_t)blockIdx.x * blockDim.x + threadIdx.x;
    if (i >= WTOT) return;
    int region; size_t o;
    if      (i < OFF110) { region = 0; o = i; }
    else if (i < OFF011) { region = 1; o = i - OFF110; }
    else if (i < OFF101) { region = 2; o = i - OFF011; }
    else                 { region = 3; o = i - OFF101; }
    const int NT = (region <= 1) ? 12 : 4;
    int e    = (int)(o & 15);
    int lane = (int)((o >> 4) & 31);
    size_t rest = o >> 9;
    int nt = (int)(rest % NT);
    int kb = (int)(rest / NT);
    int k  = kb * 32 + ((lane & 16) >> 1) + (e & 7) + ((e >> 3) << 4);
    int n  = nt * 16 + (lane & 15);
    float val;
    if (region == 0) {          // K = u*128 + v  (u,v in s-dims)
        int u = k >> 7, v = k & 127;
        val = (n < 128) ? W000a[((size_t)(u * 128 + v)) * 128 + n]
                        : W000b[((size_t)(u * 128 + v)) * 64 + (n - 128)];
    } else if (region == 1) {   // K = u*64 + v  (u,v in v-dims)
        int u = k >> 6, v = k & 63;
        val = (n < 128) ? W110a[((size_t)(u * 64 + v)) * 128 + n]
                        : W110b[((size_t)(u * 64 + v)) * 64 + (n - 128)];
    } else if (region == 2) {   // K = u*64 + v  (u: s-dim 128, v: v-dim 64)
        int u = k >> 6, v = k & 63;
        val = W011[((size_t)(u * 64 + v)) * 64 + n];
    } else {                    // K = u*128 + v (u: v-dim 64, v: s-dim 128)
        int u = k >> 7, v = k & 127;
        val = W101[((size_t)(u * 128 + v)) * 64 + n];
    }
    ws[i] = (bf16)val;
}

// ---------------------------------------------------------------------------
// Main: 256 blocks x 8 waves. Wave = (rg, ns): row-group rg (16 rows),
// N-slice ns. Stages 1-2: wave owns nt = 3ns..3ns+2 of 12. Stages 3-4:
// wave owns nt = ns (of 4) for all 3 vector components. Double-buffered
// B-fragment pipeline, one u-iteration of prefetch distance.
// ---------------------------------------------------------------------------
__global__ __launch_bounds__(256) void segate_main(
    const float* __restrict__ x, const bf16* __restrict__ ws,
    float* __restrict__ out)
{
    __shared__ bf16  sT[2][2048];      // [rg][u*16 + m], u < 128
    __shared__ bf16  vT[2][3072];      // [rg][(c*64+u)*16 + m], c<3, u<64
    __shared__ float scratch[2][3072]; // [rg][16][192]  ysc then yv
    __shared__ float gateL[2][1024];   // [rg][16][64]

    const int tid  = threadIdx.x;
    const int lane = tid & 31;
    const int wave = tid >> 5;
    const int rg   = wave & 1;         // row-group in block
    const int ns   = wave >> 1;        // N-slice 0..3
    const int m    = lane & 15;        // A: M row; B/C/D: N index
    const int half = lane >> 4;
    const int blockrow = blockIdx.x * 32;
    const int rowbase  = blockrow + rg * 16;

    // ---- cooperative staging of s/v transposes into LDS ----
    #pragma unroll
    for (int i = 0; i < 16; ++i) {                 // 4096 sT entries
        int idx = i * 256 + tid;
        int g = idx >> 11, rem = idx & 2047;
        int u = rem >> 4, mm = rem & 15;
        sT[g][rem] = (bf16)x[(size_t)(blockrow + g * 16 + mm) * XW + u];
    }
    #pragma unroll
    for (int i = 0; i < 24; ++i) {                 // 6144 vT entries
        int idx = i * 256 + tid;
        int g = (idx >= 3072) ? 1 : 0;
        int rem = idx - g * 3072;
        int mm = rem & 15, t = rem >> 4;           // t = c*64 + u
        int c = t >> 6, u = t & 63;
        vT[g][rem] = (bf16)x[(size_t)(blockrow + g * 16 + mm) * XW + 128 + u * 3 + c];
    }
    __syncthreads();

    // ---- persistent per-lane bf16 A-fragments (ISA 16-bit A 16x32 layout) ----
    const float* xr = x + (size_t)(rowbase + m) * XW;
    bf16x16 sfrag[4];           // s[m, b*32 + koff(e)]
    bf16x16 vfrag[2][3];        // v[m, b*32 + koff(e), c]
    #pragma unroll
    for (int b = 0; b < 4; ++b)
        #pragma unroll
        for (int e = 0; e < 16; ++e) {
            int kk = half * 8 + (e & 7) + ((e >> 3) << 4);
            sfrag[b][e] = (bf16)xr[b * 32 + kk];
        }
    #pragma unroll
    for (int b = 0; b < 2; ++b)
        #pragma unroll
        for (int c = 0; c < 3; ++c)
            #pragma unroll
            for (int e = 0; e < 16; ++e) {
                int kk = half * 8 + (e & 7) + ((e >> 3) << 4);
                vfrag[b][c][e] = (bf16)xr[128 + (b * 32 + kk) * 3 + c];
            }

    const bf16* sTg = sT[rg];
    const bf16* vTg = vT[rg];

    f32x8 zero = {};
    f32x8 acc[3];
    #pragma unroll
    for (int i = 0; i < 3; ++i) acc[i] = zero;

    // ---- stage 1: ysc[nt=3ns+t] += (s⊗s)·W000cat ; K = u*128+v ----
    {
        const bf16* Wp = ws + OFF000 + (size_t)(3 * ns) * 512 + (size_t)lane * 16;
        bf16x16 buf[4][3];
        #pragma unroll
        for (int b = 0; b < 4; ++b)
            #pragma unroll
            for (int t = 0; t < 3; ++t)
                buf[b][t] = *(const bf16x16*)(Wp + ((size_t)b * 12 + t) * 512);
        for (int u = 0; u < 128; ++u) {
            bf16 su = sTg[u * 16 + m];
            int un = (u < 127) ? (u + 1) : 127;
            #pragma unroll
            for (int b = 0; b < 4; ++b) {
                bf16x16 a = sfrag[b] * su;
                acc[0] = wmma_bf16(a, buf[b][0], acc[0]);
                acc[1] = wmma_bf16(a, buf[b][1], acc[1]);
                acc[2] = wmma_bf16(a, buf[b][2], acc[2]);
                const bf16* Bn = Wp + (size_t)(un * 4 + b) * 12 * 512;
                #pragma unroll
                for (int t = 0; t < 3; ++t)
                    buf[b][t] = *(const bf16x16*)(Bn + (size_t)t * 512);
            }
        }
    }
    #pragma unroll
    for (int i = 0; i < 3; ++i) acc[i] *= 1.7320508075688772f;  // PW_S00/PW_S11

    // ---- stage 2: ysc[nt=3ns+t] += (v·vT)·W110cat ; K = u*64+v ----
    {
        const bf16* Wp = ws + OFF110 + (size_t)(3 * ns) * 512 + (size_t)lane * 16;
        bf16x16 buf[2][3];
        #pragma unroll
        for (int b = 0; b < 2; ++b)
            #pragma unroll
            for (int t = 0; t < 3; ++t)
                buf[b][t] = *(const bf16x16*)(Wp + ((size_t)b * 12 + t) * 512);
        for (int u = 0; u < 64; ++u) {
            bf16 vx = vTg[(0 * 64 + u) * 16 + m];
            bf16 vy = vTg[(1 * 64 + u) * 16 + m];
            bf16 vz = vTg[(2 * 64 + u) * 16 + m];
            int un = (u < 63) ? (u + 1) : 63;
            #pragma unroll
            for (int b = 0; b < 2; ++b) {
                bf16x16 a = vfrag[b][0] * vx + vfrag[b][1] * vy + vfrag[b][2] * vz;
                acc[0] = wmma_bf16(a, buf[b][0], acc[0]);
                acc[1] = wmma_bf16(a, buf[b][1], acc[1]);
                acc[2] = wmma_bf16(a, buf[b][2], acc[2]);
                const bf16* Bn = Wp + (size_t)(un * 2 + b) * 12 * 512;
                #pragma unroll
                for (int t = 0; t < 3; ++t)
                    buf[b][t] = *(const bf16x16*)(Bn + (size_t)t * 512);
            }
        }
    }
    const float PW_S00 = 0.006987712429686844f;       // sqrt(1/20480)
    const float PW_S11 = PW_S00 * 0.5773502691896258f;

    // spill ysc tiles: row M = j + half*8, col = (3ns+t)*16 + m
    #pragma unroll
    for (int t = 0; t < 3; ++t)
        #pragma unroll
        for (int j = 0; j < 8; ++j)
            scratch[rg][(j + half * 8) * 192 + (3 * ns + t) * 16 + m] = acc[t][j] * PW_S11;
    __syncthreads();

    // ---- epilogue A: per-row std (ddof=1), SiLU cols 0..127, sigmoid gates ----
    if (lane < 4) {
        int rid = wave * 4 + lane;                 // 0..31
        int g = rid >> 4, mm = rid & 15;
        const float* ys = scratch[g] + mm * 192;
        float sum = 0.f, sq = 0.f;
        for (int w = 0; w < 192; ++w) { float t = ys[w]; sum += t; sq += t * t; }
        float mean = sum * (1.0f / 192.0f);
        float var  = (sq - 192.0f * mean * mean) * (1.0f / 191.0f);
        float inv  = 1.0f / (sqrtf(fmaxf(var, 0.0f)) + KEPS);
        float* orow = out + (size_t)(blockrow + rid) * OW;
        for (int w = 0; w < 128; ++w) {
            float t = ys[w] * inv;
            orow[w] = t / (1.0f + __expf(-t));               // silu
        }
        for (int w = 0; w < 64; ++w) {
            float t = ys[128 + w] * inv;
            gateL[g][mm * 64 + w] = 1.0f / (1.0f + __expf(-t));  // sigmoid
        }
    }
    __syncthreads();    // scratch free for yv reuse

    #pragma unroll
    for (int i = 0; i < 3; ++i) acc[i] = zero;    // acc[c] = yv[:, nt=ns, c]

    // ---- stage 3: yv_a[:,c] += (s⊗v_c)·W011 ; K = u*64+v, tile nt = ns ----
    {
        const bf16* Wp = ws + OFF011 + (size_t)ns * 512 + (size_t)lane * 16;
        bf16x16 buf0 = *(const bf16x16*)(Wp);
        bf16x16 buf1 = *(const bf16x16*)(Wp + (size_t)4 * 512);
        for (int u = 0; u < 128; ++u) {
            bf16 su = sTg[u * 16 + m];
            int un = (u < 127) ? (u + 1) : 127;
            {
                bf16x16 a0 = vfrag[0][0] * su;
                bf16x16 a1 = vfrag[0][1] * su;
                bf16x16 a2 = vfrag[0][2] * su;
                acc[0] = wmma_bf16(a0, buf0, acc[0]);
                acc[1] = wmma_bf16(a1, buf0, acc[1]);
                acc[2] = wmma_bf16(a2, buf0, acc[2]);
                buf0 = *(const bf16x16*)(Wp + (size_t)(un * 2 + 0) * 4 * 512);
            }
            {
                bf16x16 a0 = vfrag[1][0] * su;
                bf16x16 a1 = vfrag[1][1] * su;
                bf16x16 a2 = vfrag[1][2] * su;
                acc[0] = wmma_bf16(a0, buf1, acc[0]);
                acc[1] = wmma_bf16(a1, buf1, acc[1]);
                acc[2] = wmma_bf16(a2, buf1, acc[2]);
                buf1 = *(const bf16x16*)(Wp + (size_t)(un * 2 + 1) * 4 * 512);
            }
        }
    }
    // ---- stage 4: yv_b[:,c] += (v_c⊗s)·W101 ; K = u*128+v, tile nt = ns ----
    {
        const bf16* Wp = ws + OFF101 + (size_t)ns * 512 + (size_t)lane * 16;
        bf16x16 buf[4];
        #pragma unroll
        for (int b = 0; b < 4; ++b)
            buf[b] = *(const bf16x16*)(Wp + (size_t)b * 4 * 512);
        for (int u = 0; u < 64; ++u) {
            bf16 v0 = vTg[(0 * 64 + u) * 16 + m];
            bf16 v1 = vTg[(1 * 64 + u) * 16 + m];
            bf16 v2 = vTg[(2 * 64 + u) * 16 + m];
            int un = (u < 63) ? (u + 1) : 63;
            #pragma unroll
            for (int b = 0; b < 4; ++b) {
                bf16x16 a0 = sfrag[b] * v0;
                bf16x16 a1 = sfrag[b] * v1;
                bf16x16 a2 = sfrag[b] * v2;
                acc[0] = wmma_bf16(a0, buf[b], acc[0]);
                acc[1] = wmma_bf16(a1, buf[b], acc[1]);
                acc[2] = wmma_bf16(a2, buf[b], acc[2]);
                buf[b] = *(const bf16x16*)(Wp + (size_t)(un * 4 + b) * 4 * 512);
            }
        }
    }

    // spill yv (scaled by PW_V) as [M][w][c], w = ns*16 + m
    const float PW_V = 0.0078125f;
    #pragma unroll
    for (int c = 0; c < 3; ++c)
        #pragma unroll
        for (int j = 0; j < 8; ++j)
            scratch[rg][(j + half * 8) * 192 + (ns * 16 + m) * 3 + c] = acc[c][j] * PW_V;
    __syncthreads();

    // ---- epilogue B: vector-norm std (ddof=1), gate, write cols 128..319 ----
    if (lane < 4) {
        int rid = wave * 4 + lane;
        int g = rid >> 4, mm = rid & 15;
        const float* yv = scratch[g] + mm * 192;
        float sum = 0.f, sq = 0.f;
        for (int v = 0; v < 64; ++v) {
            float a0 = yv[v * 3], a1 = yv[v * 3 + 1], a2 = yv[v * 3 + 2];
            float nn = sqrtf(a0 * a0 + a1 * a1 + a2 * a2 + KEPS);
            sum += nn; sq += nn * nn;
        }
        float mean = sum * (1.0f / 64.0f);
        float var  = (sq - 64.0f * mean * mean) * (1.0f / 63.0f);
        float inv  = 1.0f / (sqrtf(fmaxf(var, 0.0f)) + KEPS);
        float* orow = out + (size_t)(blockrow + rid) * OW + 128;
        for (int v = 0; v < 64; ++v) {
            float gte = gateL[g][mm * 64 + v] * inv;
            orow[v * 3 + 0] = gte * yv[v * 3 + 0];
            orow[v * 3 + 1] = gte * yv[v * 3 + 1];
            orow[v * 3 + 2] = gte * yv[v * 3 + 2];
        }
    }
}

extern "C" void kernel_launch(void* const* d_in, const int* in_sizes, int n_in,
                              void* d_out, int out_size, void* d_ws, size_t ws_size,
                              hipStream_t stream) {
    (void)in_sizes; (void)n_in; (void)out_size; (void)ws_size;
    const float* x     = (const float*)d_in[0];
    const float* W000a = (const float*)d_in[1];
    const float* W000b = (const float*)d_in[2];
    const float* W110a = (const float*)d_in[3];
    const float* W110b = (const float*)d_in[4];
    const float* W011  = (const float*)d_in[5];
    const float* W101  = (const float*)d_in[6];
    bf16* ws = (bf16*)d_ws;

    int pblocks = (int)((WTOT + 255) / 256);
    prep_weights<<<pblocks, 256, 0, stream>>>(W000a, W000b, W110a, W110b, W011, W101, ws);

    // 8192 rows / 32 rows per block = 256 blocks, 8 waves each
    segate_main<<<256, 256, 0, stream>>>(x, ws, (float*)d_out);
}
